// BertSelfAttention_72267119723139
// MI455X (gfx1250) — matrix operand order repaired and checked by reading, not compiled, not run on hardware
//
#include <hip/hip_runtime.h>
#include <cmath>

// ---------------------------------------------------------------------------
// BERT self-attention for MI455X (gfx1250, wave32, WMMA 16x16x32 f16,
// async-to-LDS staging on ASYNCcnt).  B=4, L=2048, D=1024, H=16, DH=64.
// ---------------------------------------------------------------------------

typedef __attribute__((ext_vector_type(16))) _Float16 v16h;
typedef __attribute__((ext_vector_type(8)))  _Float16 v8h;
typedef __attribute__((ext_vector_type(8)))  float    v8f;

#define B_  4
#define L_  2048
#define D_  1024
#define H_  16
#define DH_ 64

// ---------------------------------------------------------------------------
// Fragment loaders (wave32, contiguous k: lane half h holds k = 16h + s; A and B agree).
// Work for both global and LDS source pointers (lower to b128 loads).
// ---------------------------------------------------------------------------
union HFrag { v16h v; v8h h[2]; int d[8]; };

__device__ inline v16h load_frag_a(const _Float16* __restrict__ base, int ld,
                                   int row0, int k0, int lane) {
  const int m = lane & 15, sub = lane >> 4;
  const _Float16* p = base + (size_t)(row0 + m) * ld + k0 + sub * 16;
  HFrag f;
  f.h[0] = *(const v8h*)(p);
  f.h[1] = *(const v8h*)(p + 8);
  return f.v;
}

__device__ inline v16h load_frag_bt(const _Float16* __restrict__ base, int ld,
                                    int n0, int k0, int lane) {
  const int n = lane & 15, sub = lane >> 4;
  const _Float16* p = base + (size_t)(n0 + n) * ld + k0 + sub * 16;
  HFrag f;
  f.h[0] = *(const v8h*)(p);
  f.h[1] = *(const v8h*)(p + 8);
  return f.v;
}

__device__ inline v8f wmma_f16(v16h a, v16h b, v8f c) {
  return __builtin_amdgcn_wmma_f32_16x16x32_f16(
      /*neg_a=*/false, a, /*neg_b=*/false, b,
      /*c_mod=*/(short)0, c, /*reuse_a=*/false, /*reuse_b=*/false);
}

// ---------------------------------------------------------------------------
// CDNA5 async memory->LDS copy (16B per lane, tracked on ASYNCcnt).
//   lds_off : wave-relative LDS byte address (low 32 bits of a generic
//             __shared__ pointer, per ISA flat->LDS address rules)
//   goff    : 32-bit byte offset added to the uniform SGPR base (GVS mode)
// ---------------------------------------------------------------------------
__device__ inline void async_copy_b128(unsigned lds_off, unsigned goff,
                                       const void* sbase) {
  asm volatile("global_load_async_to_lds_b128 %0, %1, %2"
               :: "v"(lds_off), "v"(goff), "s"(sbase)
               : "memory");
}
#define ASYNC_WAIT_4() asm volatile("s_wait_asynccnt 0x4" ::: "memory")
#define ASYNC_WAIT_0() asm volatile("s_wait_asynccnt 0x0" ::: "memory")

// ---------------------------------------------------------------------------
// Kernel 0: f32 -> f16 conversion (grid-stride).
// ---------------------------------------------------------------------------
__global__ __launch_bounds__(256) void k_f32_to_f16(const float* __restrict__ in,
                                                    _Float16* __restrict__ out,
                                                    size_t n) {
  size_t i = (size_t)blockIdx.x * blockDim.x + threadIdx.x;
  const size_t stride = (size_t)gridDim.x * blockDim.x;
  for (; i < n; i += stride) out[i] = (_Float16)in[i];
}

// ---------------------------------------------------------------------------
// Kernel 1/4: GEMM  C[m,n] = sum_k A[m,k] * W[n,k]
// A: [M,K] f16 row-major.  W: [N,K] f16 row-major (transposed-B layout).
// Block tile 128x128, 8 waves in 4(M) x 2(N); wave tile 32x64 = 8 WMMA accums.
//
// Data path: double-buffered async-to-LDS staging. Each K-step the block
// stages unique 128x32 A and B slabs (16 KB) ONCE via
// global_load_async_to_lds_b128 (4 wave-instructions per wave per slab pair),
// overlapped with the WMMAs of the current slab; fragments are then fed from
// LDS with ds_load_b128. This removes the 3x duplicated global fetches of the
// direct-load version and keeps the WMMA pipe fed from LDS.
// ---------------------------------------------------------------------------
template <typename OutT>
__global__ __launch_bounds__(256) void k_gemm_nt(const _Float16* __restrict__ A,
                                                 const _Float16* __restrict__ W,
                                                 OutT* __restrict__ C,
                                                 int M, int N, int K) {
  __shared__ __align__(16) _Float16 As[2][128 * 32];   // 2 x 8 KB
  __shared__ __align__(16) _Float16 Bs[2][128 * 32];   // 2 x 8 KB

  const int tid  = threadIdx.x;
  const int lane = tid & 31;
  const int wave = tid >> 5;                 // 0..7
  const int wm = (wave >> 1) * 32;           // wave row offset inside 128
  const int wn = (wave & 1) * 64;            // wave col offset inside 128
  const int m_blk = blockIdx.y * 128;
  const int n_blk = blockIdx.x * 128;

  // Cooperative copy plan: 8 KB slab = 512 x 16B chunks; 2 chunks per thread.
  const int c0 = tid, c1 = tid + 256;
  const int r0 = c0 >> 2, f0 = (c0 & 3) * 8;           // row, half-offset
  const int r1 = c1 >> 2, f1 = (c1 & 3) * 8;
  const _Float16* Abase = A + (size_t)m_blk * K;       // uniform -> SGPR pair
  const _Float16* Wbase = W + (size_t)n_blk * K;
  const unsigned ga0 = (unsigned)((r0 * K + f0) * 2);  // byte offsets
  const unsigned ga1 = (unsigned)((r1 * K + f1) * 2);
  const unsigned la0 = (unsigned)(uintptr_t)&As[0][r0 * 32 + f0];
  const unsigned la1 = (unsigned)(uintptr_t)&As[0][r1 * 32 + f1];
  const unsigned lb0 = (unsigned)(uintptr_t)&Bs[0][r0 * 32 + f0];
  const unsigned lb1 = (unsigned)(uintptr_t)&Bs[0][r1 * 32 + f1];

  // Prologue: stage slab 0 into buffer 0.
  async_copy_b128(la0, ga0, Abase);
  async_copy_b128(la1, ga1, Abase);
  async_copy_b128(lb0, ga0, Wbase);
  async_copy_b128(lb1, ga1, Wbase);

  v8f acc[2][4] = {};
  const int nk = K >> 5;

  for (int kk = 0; kk < nk; ++kk) {
    const int cur = kk & 1;
    const unsigned nb = (unsigned)((cur ^ 1) * 8192);  // next-buffer LDS bias
    if (kk + 1 < nk) {
      const unsigned kb = (unsigned)((kk + 1) << 6);   // 32 halfs = 64 bytes
      async_copy_b128(la0 + nb, ga0 + kb, Abase);
      async_copy_b128(la1 + nb, ga1 + kb, Abase);
      async_copy_b128(lb0 + nb, ga0 + kb, Wbase);
      async_copy_b128(lb1 + nb, ga1 + kb, Wbase);
      ASYNC_WAIT_4();          // in-order completion => current slab landed
    } else {
      ASYNC_WAIT_0();
    }
    __syncthreads();           // slab visible to all waves

    const _Float16* Ap = &As[cur][0];
    const _Float16* Bp = &Bs[cur][0];
    v16h a0 = load_frag_a(Ap, 32, wm,      0, lane);
    v16h a1 = load_frag_a(Ap, 32, wm + 16, 0, lane);
    v16h b0 = load_frag_bt(Bp, 32, wn,      0, lane);
    v16h b1 = load_frag_bt(Bp, 32, wn + 16, 0, lane);
    v16h b2 = load_frag_bt(Bp, 32, wn + 32, 0, lane);
    v16h b3 = load_frag_bt(Bp, 32, wn + 48, 0, lane);

    acc[0][0] = wmma_f16(a0, b0, acc[0][0]);
    acc[0][1] = wmma_f16(a0, b1, acc[0][1]);
    acc[0][2] = wmma_f16(a0, b2, acc[0][2]);
    acc[0][3] = wmma_f16(a0, b3, acc[0][3]);
    acc[1][0] = wmma_f16(a1, b0, acc[1][0]);
    acc[1][1] = wmma_f16(a1, b1, acc[1][1]);
    acc[1][2] = wmma_f16(a1, b2, acc[1][2]);
    acc[1][3] = wmma_f16(a1, b3, acc[1][3]);

    __syncthreads();           // reads done before buffer is overwritten
  }

  // C/D layout: VGPR j, lanes 0-15 -> (M=j, N=lane); lanes 16-31 -> (M=j+8).
  const int cn = lane & 15;
  const int cm = (lane >> 4) * 8;
  for (int i = 0; i < 2; ++i)
    for (int t = 0; t < 4; ++t)
      for (int j = 0; j < 8; ++j) {
        const size_t row = (size_t)(m_blk + wm + 16 * i + cm + j);
        const size_t col = (size_t)(n_blk + wn + 16 * t + cn);
        C[row * N + col] = (OutT)acc[i][t][j];
      }
}

// ---------------------------------------------------------------------------
// Kernel 2: RoPE (chunk-2 rotation) + head split.
//   qkv:  [B, L, 3D] f16   (GEMM1 output)
//   Qh,Kh: [B,H,L,DH] f16  (rotated)
//   Vt:   [B,H,DH,L] f16   (transposed: A-operand of O^T = V^T x P^T)
// ---------------------------------------------------------------------------
__global__ __launch_bounds__(256) void k_rope_split(const _Float16* __restrict__ qkv,
                                                    const float* __restrict__ sinb,
                                                    const float* __restrict__ cosb,
                                                    _Float16* __restrict__ Qh,
                                                    _Float16* __restrict__ Kh,
                                                    _Float16* __restrict__ Vt) {
  const int idx = blockIdx.x * 256 + threadIdx.x;
  const int i = idx & 31;
  const int h = (idx >> 5) & (H_ - 1);
  const int l = (idx >> 9) & (L_ - 1);
  const int b = idx >> 20;

  const _Float16* row = qkv + (size_t)(b * L_ + l) * (3 * D_);
  const float s = sinb[l * (DH_ / 2) + i];
  const float c = cosb[l * (DH_ / 2) + i];
  const int col = h * DH_ + i;

  const float q1 = (float)row[col],            q2 = (float)row[col + 32];
  const float k1 = (float)row[D_ + col],       k2 = (float)row[D_ + col + 32];
  const float v1 = (float)row[2 * D_ + col],   v2 = (float)row[2 * D_ + col + 32];

  const size_t bh = (size_t)b * H_ + h;
  _Float16* qo = Qh + (bh * L_ + l) * DH_ + i;
  qo[0]  = (_Float16)( q1 * c + q2 * s);
  qo[32] = (_Float16)(-q1 * s + q2 * c);
  _Float16* ko = Kh + (bh * L_ + l) * DH_ + i;
  ko[0]  = (_Float16)( k1 * c + k2 * s);
  ko[32] = (_Float16)(-k1 * s + k2 * c);
  Vt[(bh * DH_ + i)      * L_ + l] = (_Float16)v1;
  Vt[(bh * DH_ + i + 32) * L_ + l] = (_Float16)v2;
}

// ---------------------------------------------------------------------------
// Kernel 3: flash attention, transposed formulation (unchanged from round 2:
// compiles to max3-tree softmax + 10 bpermutes/block + 8 WMMA/block).
// ---------------------------------------------------------------------------
__global__ __launch_bounds__(256) void k_flash_attn(const _Float16* __restrict__ Qh,
                                                    const _Float16* __restrict__ Kh,
                                                    const _Float16* __restrict__ Vt,
                                                    _Float16* __restrict__ attn) {
  const int lane = threadIdx.x & 31;
  const int wave = threadIdx.x >> 5;
  const int sub  = lane >> 4;
  const int bh   = blockIdx.y;
  const int b    = bh >> 4;
  const int h    = bh & 15;
  const int qbase = blockIdx.x * 128 + wave * 16;

  const _Float16* Qp = Qh + (size_t)bh * L_ * DH_;
  const _Float16* Kp = Kh + (size_t)bh * L_ * DH_;
  const _Float16* Vp = Vt + (size_t)bh * DH_ * L_;

  const v16h bq0 = load_frag_bt(Qp, DH_, qbase, 0,  lane);
  const v16h bq1 = load_frag_bt(Qp, DH_, qbase, 32, lane);

  v8f o[4] = {};
  float mprev = -1e30f, lprev = 0.0f;
  const float scale = 0.125f;

  for (int kb = 0; kb < L_; kb += 32) {
    v8f st0 = {}, st1 = {};
    {
      v16h ak0a = load_frag_a(Kp, DH_, kb,      0,  lane);
      v16h ak0b = load_frag_a(Kp, DH_, kb,      32, lane);
      v16h ak1a = load_frag_a(Kp, DH_, kb + 16, 0,  lane);
      v16h ak1b = load_frag_a(Kp, DH_, kb + 16, 32, lane);
      st0 = wmma_f16(ak0a, bq0, st0);
      st0 = wmma_f16(ak0b, bq1, st0);
      st1 = wmma_f16(ak1a, bq0, st1);
      st1 = wmma_f16(ak1b, bq1, st1);
    }

    float xs0[8], xs1[8];
#pragma unroll
    for (int j = 0; j < 8; ++j) {
      xs0[j] = (float)st0[j] * scale;
      xs1[j] = (float)st1[j] * scale;
    }
    float mx = fmaxf(xs0[0], xs1[0]);
#pragma unroll
    for (int j = 1; j < 8; ++j) mx = fmaxf(mx, fmaxf(xs0[j], xs1[j]));
    mx = fmaxf(mx, __shfl_xor(mx, 16, 32));

    const float nm = fmaxf(mprev, mx);
    const float al = __expf(mprev - nm);

    union { v8h v; int d[4]; } g0, g1;
    float rs = 0.0f;
#pragma unroll
    for (int j = 0; j < 8; ++j) {
      const float p0 = __expf(xs0[j] - nm);
      const float p1 = __expf(xs1[j] - nm);
      rs += p0 + p1;
      g0.v[j] = (_Float16)p0;
      g1.v[j] = (_Float16)p1;
    }
    rs += __shfl_xor(rs, 16, 32);
    lprev = lprev * al + rs;
    mprev = nm;

#pragma unroll
    for (int t = 0; t < 4; ++t)
#pragma unroll
      for (int j = 0; j < 8; ++j) o[t][j] *= al;

    HFrag pf;
#pragma unroll
    for (int t = 0; t < 4; ++t) {
      const int cA = __shfl_xor(g0.d[t], 16, 32);
      const int cB = __shfl_xor(g1.d[t], 16, 32);
      pf.d[t]     = (sub == 0) ? g0.d[t] : cB;
      pf.d[4 + t] = (sub == 0) ? cA      : g1.d[t];
    }

#pragma unroll
    for (int t = 0; t < 4; ++t) {
      v16h av = load_frag_a(Vp, L_, t * 16, kb, lane);
      o[t] = wmma_f16(av, pf.v, o[t]);
    }
  }

  const float inv = 1.0f / lprev;
  const int q = qbase + (lane & 15);
  _Float16* orow = attn + ((size_t)(b * L_ + q)) * D_ + h * DH_;
#pragma unroll
  for (int t = 0; t < 4; ++t)
#pragma unroll
    for (int j = 0; j < 8; ++j) {
      const int d = t * 16 + j + 8 * sub;
      orow[d] = (_Float16)((float)o[t][j] * inv);
    }
}

// ---------------------------------------------------------------------------
// Host-side orchestration. Workspace layout (bytes), with deliberate reuse:
//   [0,   16M)  Xh   (hidden f16)      -> reused as Qh after GEMM1
//   [16M, 22M)  Wqkv f16
//   [22M, 24M)  Wo   f16
//   [24M, 40M)  Kh
//   [40M, 56M)  Vt
//   [56M,104M)  qkv  f16 [B,L,3D]      -> first 16M reused as attn f16
// ---------------------------------------------------------------------------
extern "C" void kernel_launch(void* const* d_in, const int* in_sizes, int n_in,
                              void* d_out, int out_size, void* d_ws, size_t ws_size,
                              hipStream_t stream) {
  (void)in_sizes; (void)n_in; (void)out_size; (void)ws_size;

  const float* hidden = (const float*)d_in[0];
  /* d_in[1] = mask : unused by the SDPA reference path */
  const float* sinb = (const float*)d_in[2];
  const float* cosb = (const float*)d_in[3];
  const float* Wqkv = (const float*)d_in[4];
  const float* Wo   = (const float*)d_in[5];
  float* out = (float*)d_out;

  char* ws = (char*)d_ws;
  const size_t MB = 1u << 20;
  _Float16* Xh    = (_Float16*)(ws + 0 * MB);
  _Float16* Wqkvh = (_Float16*)(ws + 16 * MB);
  _Float16* Woh   = (_Float16*)(ws + 22 * MB);
  _Float16* Kh    = (_Float16*)(ws + 24 * MB);
  _Float16* Vt    = (_Float16*)(ws + 40 * MB);
  _Float16* qkvh  = (_Float16*)(ws + 56 * MB);
  _Float16* Qh    = Xh;     // Xh dead after GEMM1
  _Float16* attnh = qkvh;   // qkv dead after rope-split

  k_f32_to_f16<<<1024, 256, 0, stream>>>(hidden, Xh, (size_t)B_ * L_ * D_);
  k_f32_to_f16<<<1024, 256, 0, stream>>>(Wqkv, Wqkvh, (size_t)3 * D_ * D_);
  k_f32_to_f16<<<256,  256, 0, stream>>>(Wo,   Woh,   (size_t)D_ * D_);

  k_gemm_nt<_Float16><<<dim3((3 * D_) / 128, (B_ * L_) / 128), 256, 0, stream>>>(
      Xh, Wqkvh, qkvh, B_ * L_, 3 * D_, D_);

  k_rope_split<<<(B_ * L_ * H_ * (DH_ / 2)) / 256, 256, 0, stream>>>(
      qkvh, sinb, cosb, Qh, Kh, Vt);

  k_flash_attn<<<dim3(L_ / 128, B_ * H_), 256, 0, stream>>>(Qh, Kh, Vt, attnh);

  k_gemm_nt<float><<<dim3(D_ / 128, (B_ * L_) / 128), 256, 0, stream>>>(
      attnh, Woh, out, B_ * L_, D_, D_);
}
